// RPN_48919677502091
// MI455X (gfx1250) — matrix operand order repaired and checked by reading, not compiled, not run on hardware
//
#include <hip/hip_runtime.h>
#include <cstdint>
#include <cstddef>

// ---------------- problem constants (fixed by reference) ----------------
#define H_    38
#define W_    50
#define M_REAL 1900            // H*W
#define M_PAD  1920            // padded to multiple of 64
#define CIN   2048
#define CMID  256
#define HP    44               // padded fmap height (>= H+2, headroom for padded M tiles)
#define WP    56               // padded fmap width  (>= W+2)
#define K1C   576              // (CIN*9)/32 k-chunks of 32
#define NA    17100            // 1900 * 9 anchors
#define NB64  268              // ceil(NA/64) 64-bit mask words
#define PSORT 32768            // next pow2 >= NA
#define NMS_TH 0.5f

typedef _Float16 v8h  __attribute__((ext_vector_type(8)));
typedef _Float16 v16h __attribute__((ext_vector_type(16)));
typedef float    v8f  __attribute__((ext_vector_type(8)));

__device__ __forceinline__ v16h cat16(v8h lo, v8h hi) {
  return __builtin_shufflevector(lo, hi, 0,1,2,3,4,5,6,7,8,9,10,11,12,13,14,15);
}
__device__ __forceinline__ v8f wmma_f16(v16h a, v16h b, v8f c) {
  // D = A(16x32 f16) * B(32x16 f16) + C(f32)
  return __builtin_amdgcn_wmma_f32_16x16x32_f16(false, a, false, b, (short)0, c, false, false);
}
// gfx1250 async copy global->LDS (ASYNCcnt path, CDNA5 ISA).
// Per-lane: 16 contiguous global bytes -> 16 contiguous LDS bytes.
__device__ __forceinline__ void async_ld_b128(uint32_t lds_byte_addr, const void* gaddr) {
  asm volatile("global_load_async_to_lds_b128 %0, %1, off"
               :: "v"(lds_byte_addr), "v"(gaddr) : "memory");
}
__device__ __forceinline__ void wait_asynccnt0() {
  asm volatile("s_wait_asynccnt 0x0" ::: "memory");
}

// ---------------- workspace layout (bytes) ----------------
constexpr size_t AL(size_t x) { return (x + 255) & ~size_t(255); }
constexpr size_t OFF_FMAPT = 0;                                   // f16 [HP][WP][CIN] zero-padded, channel-last
constexpr size_t SZ_FMAPT  = (size_t)CIN * HP * WP * 2;
constexpr size_t OFF_BMAT  = AL(OFF_FMAPT + SZ_FMAPT);            // f16 [K1C][4][64][32]
constexpr size_t SZ_BMAT   = (size_t)K1C * 4 * 64 * 32 * 2;
constexpr size_t OFF_MIDH  = AL(OFF_BMAT + SZ_BMAT);              // f16 [M_PAD][CMID]
constexpr size_t SZ_MIDH   = (size_t)M_PAD * CMID * 2;
constexpr size_t OFF_W2T   = AL(OFF_MIDH + SZ_MIDH);              // f16 [64][256]
constexpr size_t OFF_BIAS2 = AL(OFF_W2T + 64 * 256 * 2);          // f32 [64]
constexpr size_t OFF_RAW2  = AL(OFF_BIAS2 + 64 * 4);              // f32 [M_PAD][64]
constexpr size_t OFF_FG    = AL(OFF_RAW2 + (size_t)M_PAD * 64 * 4); // f32 [NA]
constexpr size_t OFF_ROIS  = AL(OFF_FG + (size_t)NA * 4);         // f32 [NA][4] clipped rois
constexpr size_t OFF_KEY   = AL(OFF_ROIS + (size_t)NA * 16);      // f32 [PSORT]
constexpr size_t OFF_SIDX  = AL(OFF_KEY + (size_t)PSORT * 4);     // i32 [PSORT]
constexpr size_t OFF_KEEPS = AL(OFF_SIDX + (size_t)PSORT * 4);    // i32 [NA]
constexpr size_t OFF_MASK  = AL(OFF_KEEPS + (size_t)NA * 4);      // u64 [NA][NB64]  (~36.7 MB)

// output sections (floats)
#define OUT_ROIS  0
#define OUT_KEEP  (NA * 4)
#define OUT_ANC   (NA * 5)
#define OUT_LOC   (NA * 9)
#define OUT_SCORE (NA * 13)

// ---------------- pre-pass kernels ----------------
// channel-last zero-padded f16 feature map:
// fmapT[(yy*WP+xx)*CIN + c] = fmap[c][yy-1][xx-1] inside image, else 0.
// Channel-last makes every A-tile row (32 consecutive channels at one spatial
// position) 64 contiguous bytes -> async-copyable with b128 lane chunks.
__global__ __launch_bounds__(256) void k_padT(const float* __restrict__ f, _Float16* __restrict__ o) {
  int i = blockIdx.x * 256 + threadIdx.x;
  if (i >= CIN * HP * WP) return;
  int c   = i & (CIN - 1);
  int pos = i >> 11;                 // CIN == 2048
  int yy = pos / WP, xx = pos % WP;
  float v = 0.f;
  if (yy >= 1 && yy <= H_ && xx >= 1 && xx <= W_)
    v = f[(size_t)c * (H_ * W_) + (yy - 1) * W_ + (xx - 1)];
  o[i] = (_Float16)v;
}

// weight swizzle: Bmat[kc][ng][nl][kl] = conv_w[n][c][tap], k = tap*2048 + c
__global__ __launch_bounds__(256) void k_w1(const float* __restrict__ w, _Float16* __restrict__ o) {
  int i = blockIdx.x * 256 + threadIdx.x;
  if (i >= K1C * 4 * 64 * 32) return;
  int kc = i >> 13;              // / (4*64*32)
  int r  = i & 8191;
  int ng = r >> 11;              // / (64*32)
  int r2 = r & 2047;
  int nl = r2 >> 5, kl = r2 & 31;
  int tap = kc >> 6;
  int c   = ((kc & 63) << 5) + kl;
  int n   = (ng << 6) + nl;
  o[i] = (_Float16)w[((size_t)n * CIN + c) * 9 + tap];
}

// combined 1x1 weights: W2t[n][k] (n<18 score, 18<=n<54 loc, else 0) + bias2[64]
__global__ __launch_bounds__(256) void k_w2(const float* __restrict__ sw, const float* __restrict__ sb,
                                            const float* __restrict__ lw, const float* __restrict__ lb,
                                            _Float16* __restrict__ w2t, float* __restrict__ bias2) {
  int i = blockIdx.x * 256 + threadIdx.x;
  if (i >= 64 * 256) return;
  int n = i >> 8, k = i & 255;
  float v = 0.f;
  if (n < 18) v = sw[n * 256 + k];
  else if (n < 54) v = lw[(n - 18) * 256 + k];
  w2t[i] = (_Float16)v;
  if (i < 64) bias2[i] = (i < 18) ? sb[i] : ((i < 54) ? lb[i - 18] : 0.f);
}

// ---------------- GEMM1: implicit-GEMM 3x3 conv, fused bias+ReLU, f16 out ----------------
// block = 256 threads (8 waves). Block tile: M=64 spatial x N=64 cout.
// Fully async-staged, double-buffered: per K-step each wave issues exactly two
// global_load_async_to_lds_b128 (A tile share + B tile share), computes on the
// other buffer, then waits + one barrier. Copy(kc+1) overlaps WMMA(kc).
__global__ __launch_bounds__(256) void k_gemm1(const _Float16* __restrict__ fpT,
                                               const _Float16* __restrict__ bmat,
                                               const float* __restrict__ bias,
                                               _Float16* __restrict__ midH) {
  __shared__ _Float16 As[2][64 * 40];   // [m][k], k-stride 40 (rows 80B, 16B aligned)
  __shared__ _Float16 Bs[2][64 * 40];   // [n][k]
  const int t    = threadIdx.x;
  const int lane = t & 31;
  const int wid  = t >> 5;
  const int wm   = wid & 3;
  const int wn   = wid >> 2;
  const int m0   = blockIdx.x * 64;
  const int ng   = blockIdx.y;

  // staging: thread t copies one 16B chunk of A and one of B per stage
  const int srow = t >> 2;            // 0..63  (m row for A, n row for B)
  const int seg  = t & 3;             // 16B segment within the 64B k-row
  const int am   = m0 + srow;
  const int ay   = am / W_;
  const int ax   = am % W_;
  const size_t apos0 = (size_t)ay * WP + ax;   // spatial position for tap (0,0)

  const uint32_t a_dst[2] = {
    (uint32_t)(uintptr_t)(&As[0][srow * 40 + seg * 8]),
    (uint32_t)(uintptr_t)(&As[1][srow * 40 + seg * 8]) };
  const uint32_t b_dst[2] = {
    (uint32_t)(uintptr_t)(&Bs[0][srow * 40 + seg * 8]),
    (uint32_t)(uintptr_t)(&Bs[1][srow * 40 + seg * 8]) };

  v8f acc0 = {}; v8f acc1 = {};
  const int hrow = lane & 15, half = lane >> 4;

  auto stage = [&](int kc, int buf) {
    const int tap = kc >> 6;
    const int c0  = (kc & 63) << 5;
    const int dy  = tap / 3, dx = tap % 3;
    const _Float16* ag = fpT + (apos0 + (size_t)dy * WP + dx) * CIN + c0 + seg * 8;
    async_ld_b128(a_dst[buf], ag);
    const _Float16* bg = bmat + (((size_t)kc * 4 + ng) << 11) + srow * 32 + seg * 8;
    async_ld_b128(b_dst[buf], bg);
  };

  stage(0, 0);
  wait_asynccnt0();
  __syncthreads();

  for (int kc = 0; kc < K1C; ++kc) {
    const int cur = kc & 1;
    if (kc + 1 < K1C) {
      stage(kc + 1, cur ^ 1);
      if (kc + 2 < K1C)   // warm B slab kc+2 into L2 (global_prefetch_b8)
        __builtin_prefetch(bmat + (((size_t)(kc + 2) * 4 + ng) << 11) + t * 8, 0, 1);
    }
    // A fragment (ISA 7.12.2): lane<16: K 0..7,16..23 ; lane>=16: K 8..15,24..31
    const _Float16* As_c = As[cur];
    const _Float16* Bs_c = Bs[cur];
    const _Float16* ap  = As_c + (wm * 16 + hrow) * 40 + half * 8;
    v16h af = cat16(*(const v8h*)ap, *(const v8h*)(ap + 16));
    // B fragments: lane holds column n=hrow, K 8h..8h+7, 16+8h..16+8h+7 (same order as A)
    const _Float16* bp0 = Bs_c + (wn * 32 + hrow) * 40 + half * 8;
    v16h bf0 = cat16(*(const v8h*)bp0, *(const v8h*)(bp0 + 16));
    const _Float16* bp1 = Bs_c + (wn * 32 + 16 + hrow) * 40 + half * 8;
    v16h bf1 = cat16(*(const v8h*)bp1, *(const v8h*)(bp1 + 16));

    acc0 = wmma_f16(af, bf0, acc0);
    acc1 = wmma_f16(af, bf1, acc1);

    wait_asynccnt0();   // own stage(kc+1) copies landed (ran during the WMMAs)
    __syncthreads();    // everyone's landed; prior reads of buf cur^1 retired
  }

  // epilogue: bias + ReLU, f16 store to midH[m][c]
  const int orb = m0 + wm * 16 + half * 8;
  const int cb  = ng * 64 + wn * 32 + hrow;
#pragma unroll
  for (int j = 0; j < 8; ++j) {
    const int orow = orb + j;
    float v0 = acc0[j] + bias[cb];
    float v1 = acc1[j] + bias[cb + 16];
    midH[(size_t)orow * CMID + cb]      = (_Float16)(v0 > 0.f ? v0 : 0.f);
    midH[(size_t)orow * CMID + cb + 16] = (_Float16)(v1 > 0.f ? v1 : 0.f);
  }
}

// ---------------- GEMM2: 1x1 convs (score+loc fused), K=256, N=64 ----------------
__global__ __launch_bounds__(256) void k_gemm2(const _Float16* __restrict__ midH,
                                               const _Float16* __restrict__ w2t,
                                               const float* __restrict__ bias2,
                                               float* __restrict__ raw2) {
  const int lane = threadIdx.x & 31, wid = threadIdx.x >> 5;
  const int mt = blockIdx.x * 8 + wid;        // 0..119
  const int m0 = mt * 16;
  const int hrow = lane & 15, half = lane >> 4;
  v8f acc[4] = {{}, {}, {}, {}};
  const _Float16* arow = midH + (size_t)(m0 + hrow) * 256;
#pragma unroll
  for (int kc = 0; kc < 8; ++kc) {
    const _Float16* ap = arow + kc * 32 + half * 8;
    v16h a = cat16(*(const v8h*)ap, *(const v8h*)(ap + 16));
#pragma unroll
    for (int nb = 0; nb < 4; ++nb) {
      const _Float16* bp = w2t + (size_t)(nb * 16 + hrow) * 256 + kc * 32 + half * 8;
      v16h b = cat16(*(const v8h*)bp, *(const v8h*)(bp + 16));
      acc[nb] = wmma_f16(a, b, acc[nb]);
    }
  }
#pragma unroll
  for (int nb = 0; nb < 4; ++nb) {
    const int col = nb * 16 + hrow;
    const float bb = bias2[col];
#pragma unroll
    for (int j = 0; j < 8; ++j) {
      const int row = m0 + half * 8 + j;
      raw2[(size_t)row * 64 + col] = acc[nb][j] + bb;
    }
  }
}

// ---------------- postprocess: anchors, softmax, loc2bbox, clip ----------------
__global__ __launch_bounds__(256) void k_post(const float* __restrict__ raw2,
                                              float* __restrict__ out,
                                              float* __restrict__ fg,
                                              float* __restrict__ rois) {
  int a = blockIdx.x * 256 + threadIdx.x;
  if (a >= NA) return;
  int mm = a / 9, j = a % 9;
  int y = mm / W_, x = mm % W_;
  int si = j / 3, ri = j % 3;
  const float scl[3] = {8.f, 16.f, 32.f};
  const float rat[3] = {0.5f, 1.f, 2.f};
  float aw = 16.f * scl[si] * sqrtf(rat[ri]);
  float ah = aw / rat[ri];
  float ax0 = x * 16.f + 8.f - 0.5f * aw;
  float ay0 = y * 16.f + 8.f - 0.5f * ah;
  float ax1 = ax0 + aw, ay1 = ay0 + ah;

  const float* rr = raw2 + (size_t)mm * 64;
  float s0 = rr[2 * j], s1 = rr[2 * j + 1];
  float mx = fmaxf(s0, s1);
  float e0 = __expf(s0 - mx), e1 = __expf(s1 - mx);
  float inv = 1.f / (e0 + e1);
  float p0 = e0 * inv, p1 = e1 * inv;

  float dxv = rr[18 + 4 * j + 0];
  float dyv = rr[18 + 4 * j + 1];
  float dwv = rr[18 + 4 * j + 2];
  float dhv = rr[18 + 4 * j + 3];
  float cx = ax0 + 0.5f * aw, cy = ay0 + 0.5f * ah;
  float ncx = dxv * aw + cx, ncy = dyv * ah + cy;
  float nw = __expf(dwv) * aw, nh = __expf(dhv) * ah;
  float r0 = fminf(fmaxf(ncx - 0.5f * nw, 0.f), (float)(W_ * 16));
  float r1 = fminf(fmaxf(ncy - 0.5f * nh, 0.f), (float)(H_ * 16));
  float r2 = fminf(fmaxf(ncx + 0.5f * nw, 0.f), (float)(W_ * 16));
  float r3 = fminf(fmaxf(ncy + 0.5f * nh, 0.f), (float)(H_ * 16));

  rois[a * 4 + 0] = r0; rois[a * 4 + 1] = r1;
  rois[a * 4 + 2] = r2; rois[a * 4 + 3] = r3;
  fg[a] = p1;
  out[OUT_ANC + a * 4 + 0] = ax0; out[OUT_ANC + a * 4 + 1] = ay0;
  out[OUT_ANC + a * 4 + 2] = ax1; out[OUT_ANC + a * 4 + 3] = ay1;
  out[OUT_LOC + a * 4 + 0] = dxv; out[OUT_LOC + a * 4 + 1] = dyv;
  out[OUT_LOC + a * 4 + 2] = dwv; out[OUT_LOC + a * 4 + 3] = dhv;
  out[OUT_SCORE + a * 2 + 0] = p0;
  out[OUT_SCORE + a * 2 + 1] = p1;
}

// ---------------- bitonic sort (descending score, stable via index tiebreak) ----------------
__global__ __launch_bounds__(256) void k_sinit(const float* __restrict__ fg,
                                               float* __restrict__ key, int* __restrict__ idx) {
  int i = blockIdx.x * 256 + threadIdx.x;
  if (i >= PSORT) return;
  key[i] = (i < NA) ? fg[i] : -__builtin_huge_valf();
  idx[i] = i;
}
__global__ __launch_bounds__(256) void k_bitonic(float* key, int* idx, int kk, int jj) {
  int i = blockIdx.x * 256 + threadIdx.x;
  int p = i ^ jj;
  if (p <= i) return;
  float ki = key[i], kp = key[p];
  int ii = idx[i], ip = idx[p];
  bool lowWorse = (ki < kp) || (ki == kp && ii > ip);
  bool desc = ((i & kk) == 0);
  if (desc == lowWorse) {
    key[i] = kp; key[p] = ki;
    idx[i] = ip; idx[p] = ii;
  }
}

// ---------------- NMS: bitmask build + sequential scan ----------------
__global__ __launch_bounds__(64) void k_mask(const float* __restrict__ rois,
                                             const int* __restrict__ sidx,
                                             unsigned long long* __restrict__ mask) {
  int r = blockIdx.y * 64 + threadIdx.x;
  int colb = blockIdx.x;
  if (r >= NA) return;
  int ir = sidx[r];
  float rx0 = rois[ir * 4 + 0], ry0 = rois[ir * 4 + 1];
  float rx1 = rois[ir * 4 + 2], ry1 = rois[ir * 4 + 3];
  float ra = (rx1 - rx0) * (ry1 - ry0);
  unsigned long long bits = 0ull;
  int cbase = colb * 64;
#pragma unroll 4
  for (int ci = 0; ci < 64; ++ci) {
    int c = cbase + ci;
    if (c >= NA || c <= r) continue;
    int ic = sidx[c];
    float cx0 = rois[ic * 4 + 0], cy0 = rois[ic * 4 + 1];
    float cx1 = rois[ic * 4 + 2], cy1 = rois[ic * 4 + 3];
    float xx0 = fmaxf(rx0, cx0), yy0 = fmaxf(ry0, cy0);
    float xx1 = fminf(rx1, cx1), yy1 = fminf(ry1, cy1);
    float inter = fmaxf(xx1 - xx0, 0.f) * fmaxf(yy1 - yy0, 0.f);
    float ca = (cx1 - cx0) * (cy1 - cy0);
    float iou = inter / (ra + ca - inter);
    if (iou > NMS_TH) bits |= (1ull << ci);
  }
  mask[(size_t)r * NB64 + colb] = bits;
}

__global__ __launch_bounds__(256) void k_scan(const unsigned long long* __restrict__ mask,
                                              int* __restrict__ keep_sorted) {
  __shared__ unsigned long long rem[NB64];
  __shared__ int cur;
  for (int w = threadIdx.x; w < NB64; w += 256) rem[w] = 0ull;
  __syncthreads();
  for (int r = 0; r < NA; ++r) {
    if (threadIdx.x == 0) {
      int kept = !((rem[r >> 6] >> (r & 63)) & 1ull);
      keep_sorted[r] = kept;
      cur = kept;
    }
    __syncthreads();
    if (cur) {
      const unsigned long long* mr = mask + (size_t)r * NB64;
      for (int w = threadIdx.x; w < NB64; w += 256) rem[w] |= mr[w];
    }
    __syncthreads();
  }
}

// ---------------- compaction: rois_out = sorted rois partitioned by keep, zero-filled ----------------
__global__ __launch_bounds__(256) void k_compact(const float* __restrict__ rois,
                                                 const int* __restrict__ sidx,
                                                 const int* __restrict__ keep_sorted,
                                                 float* __restrict__ out) {
  __shared__ int sf[256];
  __shared__ int base;
  if (threadIdx.x == 0) base = 0;
  __syncthreads();
  for (int start = 0; start < NA; start += 256) {
    int r = start + threadIdx.x;
    int f = (r < NA) ? keep_sorted[r] : 0;
    sf[threadIdx.x] = f;
    __syncthreads();
    for (int off = 1; off < 256; off <<= 1) {
      int v = (threadIdx.x >= off) ? sf[threadIdx.x - off] : 0;
      __syncthreads();
      sf[threadIdx.x] += v;
      __syncthreads();
    }
    if (f) {
      int pos = base + sf[threadIdx.x] - 1;
      int ir = sidx[r];
      out[OUT_ROIS + pos * 4 + 0] = rois[ir * 4 + 0];
      out[OUT_ROIS + pos * 4 + 1] = rois[ir * 4 + 1];
      out[OUT_ROIS + pos * 4 + 2] = rois[ir * 4 + 2];
      out[OUT_ROIS + pos * 4 + 3] = rois[ir * 4 + 3];
      out[OUT_KEEP + pos] = 1.0f;
    }
    __syncthreads();
    if (threadIdx.x == 255) base += sf[255];
    __syncthreads();
  }
  for (int p = base + threadIdx.x; p < NA; p += 256) {
    out[OUT_ROIS + p * 4 + 0] = 0.f;
    out[OUT_ROIS + p * 4 + 1] = 0.f;
    out[OUT_ROIS + p * 4 + 2] = 0.f;
    out[OUT_ROIS + p * 4 + 3] = 0.f;
    out[OUT_KEEP + p] = 0.f;
  }
}

// ---------------- launch ----------------
extern "C" void kernel_launch(void* const* d_in, const int* in_sizes, int n_in,
                              void* d_out, int out_size, void* d_ws, size_t ws_size,
                              hipStream_t stream) {
  const float* fmap   = (const float*)d_in[0];
  // d_in[1] = feature_stride (int, ==16, baked into constants)
  const float* conv_w = (const float*)d_in[2];
  const float* conv_b = (const float*)d_in[3];
  const float* score_w = (const float*)d_in[4];
  const float* score_b = (const float*)d_in[5];
  const float* loc_w  = (const float*)d_in[6];
  const float* loc_b  = (const float*)d_in[7];
  float* out = (float*)d_out;

  char* ws = (char*)d_ws;
  _Float16* fmapT = (_Float16*)(ws + OFF_FMAPT);
  _Float16* bmat  = (_Float16*)(ws + OFF_BMAT);
  _Float16* midH  = (_Float16*)(ws + OFF_MIDH);
  _Float16* w2t   = (_Float16*)(ws + OFF_W2T);
  float*    bias2 = (float*)(ws + OFF_BIAS2);
  float*    raw2  = (float*)(ws + OFF_RAW2);
  float*    fg    = (float*)(ws + OFF_FG);
  float*    rois  = (float*)(ws + OFF_ROIS);
  float*    key   = (float*)(ws + OFF_KEY);
  int*      sidx  = (int*)(ws + OFF_SIDX);
  int*      keeps = (int*)(ws + OFF_KEEPS);
  unsigned long long* mask = (unsigned long long*)(ws + OFF_MASK);

  // pre-pass
  k_padT<<<(CIN * HP * WP + 255) / 256, 256, 0, stream>>>(fmap, fmapT);
  k_w1<<<(K1C * 4 * 64 * 32 + 255) / 256, 256, 0, stream>>>(conv_w, bmat);
  k_w2<<<(64 * 256 + 255) / 256, 256, 0, stream>>>(score_w, score_b, loc_w, loc_b, w2t, bias2);

  // GEMMs
  k_gemm1<<<dim3(M_PAD / 64, CMID / 64), 256, 0, stream>>>(fmapT, bmat, conv_b, midH);
  k_gemm2<<<M_PAD / (16 * 8), 256, 0, stream>>>(midH, w2t, bias2, raw2);

  // decode
  k_post<<<(NA + 255) / 256, 256, 0, stream>>>(raw2, out, fg, rois);

  // sort (descending by fg score)
  k_sinit<<<PSORT / 256, 256, 0, stream>>>(fg, key, sidx);
  for (int kk = 2; kk <= PSORT; kk <<= 1)
    for (int jj = kk >> 1; jj > 0; jj >>= 1)
      k_bitonic<<<PSORT / 256, 256, 0, stream>>>(key, sidx, kk, jj);

  // NMS
  k_mask<<<dim3(NB64, NB64), 64, 0, stream>>>(rois, sidx, mask);
  k_scan<<<1, 256, 0, stream>>>(mask, keeps);

  // compaction -> rois_out + keep sections of d_out
  k_compact<<<1, 256, 0, stream>>>(rois, sidx, keeps, out);

  (void)in_sizes; (void)n_in; (void)out_size; (void)ws_size;
}
